// MPN_16269336117573
// MI455X (gfx1250) — matrix operand order repaired and checked by reading, not compiled, not run on hardware
//
#include <hip/hip_runtime.h>
#include <hip/hip_bf16.h>
#include <stdint.h>

// ---------------- problem constants (from reference) ----------------
constexpr int NA    = 100000;   // atoms
constexpr int NB    = 200000;   // bonds
constexpr int NMOLS = 4000;
constexpr int H     = 300;      // hidden
constexpr int HP    = 320;      // hidden padded to 20 * 16
constexpr int NT    = 20;       // N tiles of 16 (HP/16)
constexpr int FA    = 39;       // atom feature dim
constexpr int FB    = 50;       // bond feature dim (39+11)
constexpr int FBP   = 64;       // padded to 2 * 32
constexpr int AIN   = 339;      // 39 + 300
constexpr int AINP  = 352;      // padded to 11 * 32
constexpr int DEPTH = 6;

typedef __attribute__((ext_vector_type(16))) __bf16 v16bf;
typedef __attribute__((ext_vector_type(8)))  __bf16 bf16x8;
typedef __attribute__((ext_vector_type(8)))  float  v8f;
typedef __attribute__((ext_vector_type(4)))  unsigned int v4u;
typedef __attribute__((ext_vector_type(8)))  int v8i;
typedef __attribute__((ext_vector_type(4)))  int v4i;

// ------------------------------------------------------------------
// Pack f32 weight [Korig x Norig] (row-major, stored [in,out]) into the
// per-lane B-fragment order this code uses (contiguous k, as the A loads) for v_wmma_f32_16x16x32_bf16:
//   fragment (kt, nt): lane l, element e -> B[kt*32 + (l/16)*16 + e][nt*16 + l%16]
// Layout: dst[((kt*NT + nt)*32 + l)*16 + e], zero-padded.
// ------------------------------------------------------------------
__global__ void pack_w(const float* __restrict__ W, int Korig, int Norig,
                       __bf16* __restrict__ dst, int KT)
{
    int idx = blockIdx.x * blockDim.x + threadIdx.x;
    int total = KT * NT * 32;
    if (idx >= total) return;
    int r  = idx % (NT * 32);
    int kt = idx / (NT * 32);
    int nt = r >> 5;
    int l  = r & 31;
    int n  = nt * 16 + (l & 15);
    int kb = kt * 32 + (l >> 4) * 16;
    __bf16* d = dst + (size_t)idx * 16;
#pragma unroll
    for (int e = 0; e < 16; ++e) {
        int k = kb + e;
        float v = (k < Korig && n < Norig) ? W[(size_t)k * Norig + n] : 0.0f;
        d[e] = (__bf16)v;
    }
}

// f32 [rows x scols] -> bf16 [rows x dcols] zero-padded
__global__ void cvt_pad_bf16(const float* __restrict__ src, int scols,
                             __bf16* __restrict__ dst, int dcols, int rows)
{
    int idx = blockIdx.x * blockDim.x + threadIdx.x;
    int total = rows * dcols;
    if (idx >= total) return;
    int row = idx / dcols, c = idx % dcols;
    float v = (c < scols) ? src[(size_t)row * scols + c] : 0.0f;
    dst[(size_t)row * dcols + c] = (__bf16)v;
}

// ------------------------------------------------------------------
// WMMA GEMM: C[M x 320] = A[M x (KT*32)] @ Bpacked, f32 accumulate.
// 8 waves / block, each wave computes a 16 x 320 strip.
// Packed B is DMA'd into LDS by the Tensor Data Mover (TENSOR_LOAD_TO_LDS),
// issued by wave 0; completion via s_wait_tensorcnt + workgroup barrier.
// Inner loop: double-buffered A (global) and B (LDS) fragments feeding
// v_wmma_f32_16x16x32_bf16 with f32-resident accumulators.
// Epilogue: +addsrc (residual), +bias, optional pre-relu f32 store,
// relu, optional bf16 store.
// ------------------------------------------------------------------
__global__ void __launch_bounds__(256)
wmma_gemm(const __bf16* __restrict__ A, int lda,
          const __bf16* __restrict__ Bp, int KT,
          const float* __restrict__ addsrc,
          const float* __restrict__ bias, int nbias,
          float* __restrict__ pre_out,
          __bf16* __restrict__ bf_out,
          int M)
{
    extern __shared__ __bf16 ldsB[];   // only LDS allocation -> offset 0
    const int tid = threadIdx.x;

    // ---- TDM: DMA packed B (KT*NT*1024 bytes) from global into LDS ----
    if (tid < 32) {                    // wave 0 issues; TDM ignores EXEC
        uint64_t ga = (uint64_t)(uintptr_t)Bp;
        unsigned int nelem8 = (unsigned int)(KT * NT * 128);  // 8-byte elements
        v4u g0;
        g0[0] = 1u;                                    // count=1 (valid), load
        g0[1] = 0u;                                    // lds_addr = 0
        g0[2] = (unsigned int)ga;                      // global_addr[31:0]
        g0[3] = (unsigned int)((ga >> 32) & 0x1FFFFFFu) | (2u << 30); // [56:32], type=2
        v8i g1;
        g1[0] = (int)(3u << 16);                       // data_size=3 (8B), mask=0
        g1[1] = (int)((nelem8 & 0xFFFFu) << 16);       // tensor_dim0[15:0]
        g1[2] = (int)(((nelem8 >> 16) & 0xFFFFu) | (1u << 16)); // dim0 hi, tensor_dim1=1
        g1[3] = (int)((nelem8 & 0xFFFFu) << 16);       // tile_dim0 (<= 28160 < 2^16)
        g1[4] = 1;                                     // tile_dim1 = 1
        g1[5] = (int)nelem8;                           // tensor_dim0_stride lo
        g1[6] = 0;
        g1[7] = 0;
        v4i z4 = {0, 0, 0, 0};
        v8i z8 = {0, 0, 0, 0, 0, 0, 0, 0};
        __builtin_amdgcn_tensor_load_to_lds(g0, g1, z4, z4, z8, 0);
        __builtin_amdgcn_s_wait_tensorcnt(0);
    }
    __syncthreads();

    const int wave  = tid >> 5;
    const int lane  = tid & 31;
    const int mbase = (blockIdx.x * 8 + wave) * 16;
    const int kg    = lane >> 4;       // K-group (A layout)
    int mload = mbase + (lane & 15);
    if (mload >= M) mload = M - 1;     // clamp loads; stores predicated below

    v8f acc[NT];
#pragma unroll
    for (int nt = 0; nt < NT; ++nt)
#pragma unroll
        for (int i = 0; i < 8; ++i) acc[nt][i] = 0.0f;

    const __bf16* arow = A + (size_t)mload * lda;
    union AF { v16bf v; bf16x8 h[2]; };

    // A fragment double buffer
    AF acur, anext;
    acur.v = *(const v16bf*)(arow + kg * 16);

    for (int kt = 0; kt < KT; ++kt) {
        if (kt + 1 < KT) {             // prefetch next A fragment (global)
            anext.v = *(const v16bf*)(arow + (kt + 1) * 32 + kg * 16);
        } else {
            anext = acur;
        }
        const __bf16* bb = ldsB + (size_t)(kt * NT) * 512 + lane * 16;
        // B fragment double buffer (LDS)
        v16bf bcur = *(const v16bf*)(bb);
#pragma unroll
        for (int nt = 0; nt < NT; ++nt) {
            v16bf bnext = (nt + 1 < NT) ? *(const v16bf*)(bb + (nt + 1) * 512) : bcur;
            acc[nt] = __builtin_amdgcn_wmma_f32_16x16x32_bf16(
                false, acur.v, false, bcur, (short)0, acc[nt], false, false);
            bcur = bnext;
        }
        acur = anext;
    }

    // epilogue: C layout -> VGPR i: row = mbase + (lane/16)*8 + i, col = nt*16 + lane%16
    const int coll  = lane & 15;
    const int rbase = mbase + kg * 8;
#pragma unroll
    for (int nt = 0; nt < NT; ++nt) {
#pragma unroll
        for (int i = 0; i < 8; ++i) {
            int row = rbase + i;
            if (row >= M) continue;
            int col = nt * 16 + coll;
            float v = acc[nt][i];
            if (addsrc) v += addsrc[(size_t)row * HP + col];
            if (bias && col < nbias) v += bias[col];
            if (pre_out) pre_out[(size_t)row * HP + col] = v;
            v = fmaxf(v, 0.0f);
            if (bf_out) bf_out[(size_t)row * HP + col] = (__bf16)v;
        }
    }
}

// nei[row,:] = sum_{j<6} message[bgraph[row][j], :]   (f32 sum, bf16 store)
__global__ void gather_sum_bf16(const __bf16* __restrict__ msg,
                                const int* __restrict__ graph,
                                __bf16* __restrict__ out, int rows)
{
    constexpr int CH = HP / 8;  // 40 chunks of 8 cols
    int idx = blockIdx.x * blockDim.x + threadIdx.x;
    if (idx >= rows * CH) return;
    int row = idx / CH;
    int c0  = (idx % CH) * 8;
    float s[8];
#pragma unroll
    for (int e = 0; e < 8; ++e) s[e] = 0.0f;
    const int* g = graph + (size_t)row * 6;
#pragma unroll
    for (int j = 0; j < 6; ++j) {
        int nb = g[j];
        bf16x8 v = *(const bf16x8*)(msg + (size_t)nb * HP + c0);
#pragma unroll
        for (int e = 0; e < 8; ++e) s[e] += (float)v[e];
    }
    bf16x8 o;
#pragma unroll
    for (int e = 0; e < 8; ++e) o[e] = (__bf16)s[e];
    *(bf16x8*)(out + (size_t)row * HP + c0) = o;
}

// ainput cols 0..38 <- bf16(fatoms); cols 339..351 <- 0
__global__ void build_ainput(const float* __restrict__ fatoms,
                             __bf16* __restrict__ ain, int rows)
{
    constexpr int C = FA + (AINP - AIN);  // 39 + 13 = 52 cols handled here
    int idx = blockIdx.x * blockDim.x + threadIdx.x;
    if (idx >= rows * C) return;
    int row = idx / C, c = idx % C;
    if (c < FA)
        ain[(size_t)row * AINP + c] = (__bf16)fatoms[(size_t)row * FA + c];
    else
        ain[(size_t)row * AINP + (AIN + (c - FA))] = (__bf16)0.0f;
}

// ainput cols 39..338 <- sum_{j<6} message[agraph[row][j], :300]
__global__ void gather_sum_atoms(const __bf16* __restrict__ msg,
                                 const int* __restrict__ graph,
                                 __bf16* __restrict__ ain, int rows)
{
    constexpr int CH = 38;  // ceil(300/8) chunks (msg padded cols are zero -> safe to load)
    int idx = blockIdx.x * blockDim.x + threadIdx.x;
    if (idx >= rows * CH) return;
    int row = idx / CH;
    int c0  = (idx % CH) * 8;
    float s[8];
#pragma unroll
    for (int e = 0; e < 8; ++e) s[e] = 0.0f;
    const int* g = graph + (size_t)row * 6;
#pragma unroll
    for (int j = 0; j < 6; ++j) {
        int nb = g[j];
        bf16x8 v = *(const bf16x8*)(msg + (size_t)nb * HP + c0);
#pragma unroll
        for (int e = 0; e < 8; ++e) s[e] += (float)v[e];
    }
#pragma unroll
    for (int e = 0; e < 8; ++e) {
        int c = c0 + e;
        if (c < H) ain[(size_t)row * AINP + FA + c] = (__bf16)s[e];
    }
}

// Deterministic per-molecule mean (mol_idx sorted -> binary-search segment)
__global__ void pool_mean(const __bf16* __restrict__ atomh,
                          const int* __restrict__ molidx,
                          float* __restrict__ out, int natoms)
{
    int m   = blockIdx.x;
    int col = threadIdx.x;   // 320 threads; 300 active
    int a = 0, b = natoms;
    while (a < b) { int mid = (a + b) >> 1; if (molidx[mid] < m) a = mid + 1; else b = mid; }
    int lo = a;
    a = lo; b = natoms;
    while (a < b) { int mid = (a + b) >> 1; if (molidx[mid] < m + 1) a = mid + 1; else b = mid; }
    int hi = a;
    if (col >= H) return;
    float s = 0.0f;
    for (int i = lo; i < hi; ++i) s += (float)atomh[(size_t)i * HP + col];
    int cnt = hi - lo;
    out[(size_t)m * H + col] = s / (float)(cnt > 0 ? cnt : 1);
}

// ------------------------------------------------------------------
extern "C" void kernel_launch(void* const* d_in, const int* in_sizes, int n_in,
                              void* d_out, int out_size, void* d_ws, size_t ws_size,
                              hipStream_t stream)
{
    const float* fatoms = (const float*)d_in[0];
    const float* fbonds = (const float*)d_in[1];
    const int*   agraph = (const int*)d_in[2];
    const int*   bgraph = (const int*)d_in[3];
    const int*   molidx = (const int*)d_in[4];
    // d_in[5] = n_mols (constant 4000 per reference)
    const float* W_i = (const float*)d_in[6];
    const float* W_h = (const float*)d_in[7];
    const float* W_o = (const float*)d_in[8];
    const float* b_o = (const float*)d_in[9];
    float* out = (float*)d_out;

    // workspace carving (256B aligned)
    char* ws = (char*)d_ws;
    size_t off = 0;
    auto take = [&](size_t bytes) -> char* {
        char* p = ws + off;
        off = (off + bytes + 255) & ~(size_t)255;
        return p;
    };
    float*  binput  = (float*) take((size_t)NB * HP * 4);   // 256 MB f32 residual
    __bf16* message = (__bf16*)take((size_t)NB * HP * 2);   // 128 MB (mostly L2-resident)
    __bf16* nei     = (__bf16*)take((size_t)NB * HP * 2);   // 128 MB
    __bf16* fb16    = (__bf16*)take((size_t)NB * FBP * 2);  // 25.6 MB
    __bf16* ain     = (__bf16*)take((size_t)NA * AINP * 2); // 70.4 MB
    __bf16* atomh   = (__bf16*)take((size_t)NA * HP * 2);   // 64 MB
    __bf16* pWi     = (__bf16*)take((size_t)2  * NT * 512 * 2);
    __bf16* pWh     = (__bf16*)take((size_t)10 * NT * 512 * 2);
    __bf16* pWo     = (__bf16*)take((size_t)11 * NT * 512 * 2);

    // 1) pack weights into WMMA fragment order (tiny, once per call)
    pack_w<<<(2  * NT * 32 + 255) / 256, 256, 0, stream>>>(W_i, FB,  H, pWi, 2);
    pack_w<<<(10 * NT * 32 + 255) / 256, 256, 0, stream>>>(W_h, H,   H, pWh, 10);
    pack_w<<<(11 * NT * 32 + 255) / 256, 256, 0, stream>>>(W_o, AIN, H, pWo, 11);

    // 2) fbonds -> bf16 padded
    cvt_pad_bf16<<<(NB * FBP + 255) / 256, 256, 0, stream>>>(fbonds, FB, fb16, FBP, NB);

    // 3) binput = fbonds @ W_i (f32 store), message = relu(binput) (bf16 store)
    const int gbB = (NB + 127) / 128;  // 8 waves * 16 rows per block
    wmma_gemm<<<gbB, 256, 2 * NT * 512 * 2, stream>>>(
        fb16, FBP, pWi, 2, nullptr, nullptr, 0, binput, message, NB);

    // 4) message-passing loop: nei = gather6(message); message = relu(binput + nei @ W_h)
    for (int d = 0; d < DEPTH - 1; ++d) {
        gather_sum_bf16<<<(NB * (HP / 8) + 255) / 256, 256, 0, stream>>>(
            message, bgraph, nei, NB);
        wmma_gemm<<<gbB, 256, 10 * NT * 512 * 2, stream>>>(
            nei, HP, pWh, 10, binput, nullptr, 0, nullptr, message, NB);
    }

    // 5) ainput = [fatoms | gather6_atoms(message)] (bf16, padded to 352)
    build_ainput<<<(NA * 52 + 255) / 256, 256, 0, stream>>>(fatoms, ain, NA);
    gather_sum_atoms<<<(NA * 38 + 255) / 256, 256, 0, stream>>>(message, agraph, ain, NA);

    // 6) atom_hiddens = relu(ainput @ W_o + b_o)  (bf16 store)
    const int gbA = (NA + 127) / 128;
    wmma_gemm<<<gbA, 256, 11 * NT * 512 * 2, stream>>>(
        ain, AINP, pWo, 11, nullptr, b_o, H, nullptr, atomh, NA);

    // 7) deterministic per-molecule mean pooling
    pool_mean<<<NMOLS, HP, 0, stream>>>(atomh, molidx, out, NA);
}
